// ALIGNNEncoder_2594160247296
// MI455X (gfx1250) — matrix operand order repaired and checked by reading, not compiled, not run on hardware
//
#include <hip/hip_runtime.h>

#define NN 10000
#define EE 80000
#define TT 240000
#define BB 32
#define HF 256

typedef __attribute__((ext_vector_type(16))) __bf16 v16bf;
typedef __attribute__((ext_vector_type(8)))  float  v8f;

__device__ __forceinline__ unsigned short f2bf(float f) {
    unsigned u = __float_as_uint(f);
    u += 0x7fffu + ((u >> 16) & 1u);   // round-to-nearest-even
    return (unsigned short)(u >> 16);
}

__device__ __forceinline__ float wave_sum32(float v) {
    #pragma unroll
    for (int o = 16; o > 0; o >>= 1) v += __shfl_xor(v, o, 32);
    return v;
}

// ---------------- prep kernels ----------------

// conv_W [12][5][256][256] (W[k][n]) -> wcat_t [12][1024][256] bf16 (slots 0,1,3,4 transposed)
//                                      w2t    [12][256][256]  bf16 (slot 2 transposed)
__global__ void k_prep_convw(const float* __restrict__ convW,
                             unsigned short* __restrict__ wcat,
                             unsigned short* __restrict__ w2t) {
    long i = (long)blockIdx.x * 256 + threadIdx.x;
    long total = 12L * 5 * 256 * 256;
    if (i >= total) return;
    int n = (int)(i & 255);
    long r = i >> 8;
    int k = (int)(r & 255); r >>= 8;
    int g = (int)(r % 5);
    int layer = (int)(r / 5);
    unsigned short w = f2bf(convW[i]);
    if (g == 2) {
        w2t[((long)layer * 256 + n) * 256 + k] = w;
    } else {
        int slot = (g < 2) ? g : (g - 1);          // 0,1,3,4 -> 0,1,2,3
        wcat[((long)layer * 1024 + slot * 256 + n) * 256 + k] = w;
    }
}

// conv_b [12][5][256] -> bcat [12][1024] (order b0,b1,b3,b4)
__global__ void k_prep_bcat(const float* __restrict__ convb, float* __restrict__ bcat) {
    int i = blockIdx.x * 256 + threadIdx.x;
    if (i >= 12 * 1024) return;
    int layer = i >> 10;
    int slot  = (i >> 8) & 3;
    int n     = i & 255;
    int g = (slot < 2) ? slot : (slot + 1);
    bcat[i] = convb[((long)layer * 5 + g) * 256 + n];
}

// src [Ks][256] f32 -> dst [256][Kp] bf16 (transpose, zero-pad K)
__global__ void k_trans_pad(const float* __restrict__ src, unsigned short* __restrict__ dst,
                            int Ks, int Kp) {
    int i = blockIdx.x * 256 + threadIdx.x;
    if (i >= 256 * Kp) return;
    int n = i / Kp, k = i % Kp;
    dst[i] = (k < Ks) ? f2bf(src[(long)k * 256 + n]) : (unsigned short)0;
}

// atom_features [N][92] -> bf16 [N][96] (pad)
__global__ void k_pad_atoms(const float* __restrict__ af, unsigned short* __restrict__ out, int n) {
    long i = (long)blockIdx.x * 256 + threadIdx.x;
    if (i >= (long)n * 96) return;
    int r = (int)(i / 96), k = (int)(i % 96);
    out[i] = (k < 92) ? f2bf(af[(long)r * 92 + k]) : (unsigned short)0;
}

__global__ void k_zero(float* __restrict__ p, long n) {
    long i = (long)blockIdx.x * 256 + threadIdx.x;
    long stride = (long)gridDim.x * 256;
    for (; i < n; i += stride) p[i] = 0.0f;
}

// ---------------- RBF + MLP1 (Linear->LN->SiLU), 64 outputs, fused per row ----------------
__global__ void __launch_bounds__(128)
k_rbf_mlp(const float* __restrict__ xin, unsigned short* __restrict__ out,
          const float* __restrict__ W1, const float* __restrict__ b1,
          const float* __restrict__ ln, int rows, int bins, float vmin, float vmax) {
    extern __shared__ float w1s[];                  // bins*64
    __shared__ float b1s[64], lns[128];
    for (int t = threadIdx.x; t < bins * 64; t += blockDim.x) w1s[t] = W1[t];
    if (threadIdx.x < 64)  b1s[threadIdx.x] = b1[threadIdx.x];
    if (threadIdx.x < 128) lns[threadIdx.x] = ln[threadIdx.x];
    __syncthreads();
    int r = blockIdx.x * blockDim.x + threadIdx.x;
    if (r >= rows) return;
    float xv = xin[r];
    float step  = (vmax - vmin) / (float)bins;
    float gamma = 0.5f / (step * step);
    float cstep = (vmax - vmin) / (float)(bins - 1);
    float acc[64];
    #pragma unroll
    for (int j = 0; j < 64; j++) acc[j] = b1s[j];
    for (int kk = 0; kk < bins; kk++) {
        float d  = xv - (vmin + kk * cstep);
        float rv = __expf(-gamma * d * d);
        const float* wr = &w1s[kk * 64];
        #pragma unroll
        for (int j = 0; j < 64; j++) acc[j] = fmaf(rv, wr[j], acc[j]);
    }
    float m = 0.f;
    #pragma unroll
    for (int j = 0; j < 64; j++) m += acc[j];
    m *= (1.f / 64.f);
    float q = 0.f;
    #pragma unroll
    for (int j = 0; j < 64; j++) { float d = acc[j] - m; q += d * d; }
    float rstd = rsqrtf(q * (1.f / 64.f) + 1e-5f);
    #pragma unroll
    for (int j = 0; j < 64; j++) {
        float t = (acc[j] - m) * rstd * lns[j] + lns[64 + j];
        t = t / (1.f + __expf(-t));                 // SiLU
        out[(long)r * 64 + j] = f2bf(t);
    }
}

// ---------------- bf16 WMMA GEMM: C[rows,Nc] = A[rows,K] @ Wt[Nc,K]^T + bias ----------------
// 8 waves/block; block computes 16 rows x 256 cols (2 N-tiles per wave).
// A panel (16 x K bf16) staged once in LDS; inner loop: ds_load_b128 + global_load_b128 + 2 WMMA.
// K % 32 == 0, Nc % 256 == 0.
template <int K>
__global__ void __launch_bounds__(256)
k_gemm(const unsigned short* __restrict__ A, const unsigned short* __restrict__ Wt,
       const float* __restrict__ bias, float* __restrict__ C, int rows, int Nc) {
    constexpr int KW  = K / 2;                      // dwords per A row
    constexpr int AST = KW + 4;                     // padded LDS row stride (bank spread)
    __shared__ unsigned As[16 * AST];
    int row0 = blockIdx.x * 16;
    // ---- stage A panel once (vectorized, zero-pad OOB rows) ----
    for (int t = threadIdx.x * 4; t < 16 * KW; t += 1024) {
        int m = t / KW, kq = t % KW;
        int row = row0 + m;
        uint4 v = {0u, 0u, 0u, 0u};
        if (row < rows) v = *(const uint4*)(A + (size_t)row * K + 2 * kq);
        *(uint4*)&As[m * AST + kq] = v;
    }
    __syncthreads();
    int wv = threadIdx.x >> 5, lane = threadIdx.x & 31;
    int half = lane >> 4, mn = lane & 15;
    int n0 = blockIdx.y * 256 + wv * 16;
    v8f acc0 = {}, acc1 = {};
    const unsigned* bp0 = (const unsigned*)(Wt + (size_t)(n0 + mn) * K);
    const unsigned* bp1 = (const unsigned*)(Wt + (size_t)(n0 + 128 + mn) * K);
    #pragma unroll
    for (int k0 = 0; k0 < K; k0 += 32) {
        union { v16bf v; unsigned u[8]; } a, b0, b1;
        // A 16x32 bf16 fragment: lane (half,mn): k = 16*half + i
        #pragma unroll
        for (int i = 0; i < 8; i++)
            a.u[i] = As[mn * AST + k0 / 2 + 8 * half + i];
        // B 32x16 bf16 fragments: lane (half,n): k = 16*half + i, contiguous in Wt[n][k]
        #pragma unroll
        for (int i = 0; i < 8; i++) b0.u[i] = bp0[k0 / 2 + 8 * half + i];
        #pragma unroll
        for (int i = 0; i < 8; i++) b1.u[i] = bp1[k0 / 2 + 8 * half + i];
        acc0 = __builtin_amdgcn_wmma_f32_16x16x32_bf16(false, a.v, false, b0.v,
                                                       (short)0, acc0, false, false);
        acc1 = __builtin_amdgcn_wmma_f32_16x16x32_bf16(false, a.v, false, b1.v,
                                                       (short)0, acc1, false, false);
    }
    float bv0 = bias ? bias[n0 + mn] : 0.0f;
    float bv1 = bias ? bias[n0 + 128 + mn] : 0.0f;
    #pragma unroll
    for (int r = 0; r < 8; r++) {                   // C: lane n=mn, M = r + 8*half
        int row = row0 + r + 8 * half;
        if (row < rows) {
            C[(size_t)row * Nc + n0 + mn]       = acc0[r] + bv0;
            C[(size_t)row * Nc + n0 + 128 + mn] = acc1[r] + bv1;
        }
    }
}

// ---------------- LayerNorm (+optional SiLU) over 256, wave per row, f32 + bf16 mirror ----------------
__global__ void __launch_bounds__(256)
k_ln_act(float* __restrict__ X, unsigned short* __restrict__ Xh,
         const float* __restrict__ lns, const float* __restrict__ lnb, int rows, int do_silu) {
    int r = blockIdx.x * 8 + (threadIdx.x >> 5);
    if (r >= rows) return;
    int lane = threadIdx.x & 31;
    float v[8];
    #pragma unroll
    for (int j = 0; j < 8; j++) v[j] = X[(size_t)r * 256 + lane + 32 * j];
    float s = 0.f;
    #pragma unroll
    for (int j = 0; j < 8; j++) s += v[j];
    float m = wave_sum32(s) * (1.f / 256.f);
    float q = 0.f;
    #pragma unroll
    for (int j = 0; j < 8; j++) { float d = v[j] - m; q += d * d; }
    float rstd = rsqrtf(wave_sum32(q) * (1.f / 256.f) + 1e-5f);
    #pragma unroll
    for (int j = 0; j < 8; j++) {
        int c = lane + 32 * j;
        float t = (v[j] - m) * rstd * lns[c] + lnb[c];
        if (do_silu) t = t / (1.f + __expf(-t));
        X[(size_t)r * 256 + c]  = t;
        Xh[(size_t)r * 256 + c] = f2bf(t);
    }
}

// ---------------- EGGC edge stage: sigma, e_out = LN(sigma*e), atomic scatter ----------------
__global__ void __launch_bounds__(256)
k_edge(const int* __restrict__ src, const int* __restrict__ dst,
       const float* __restrict__ Hout, const float* __restrict__ Eg,
       float* __restrict__ Ef, unsigned short* __restrict__ Eh,
       float* __restrict__ ssh, float* __restrict__ ss,
       const float* __restrict__ lns, const float* __restrict__ lnb, int ne) {
    int t = blockIdx.x * 8 + (threadIdx.x >> 5);
    if (t >= ne) return;
    int lane = threadIdx.x & 31;
    long s = src[t], d = dst[t];
    const float* hs = Hout + s * 1024;              // e_src = h@W0+b0
    const float* hd = Hout + d * 1024 + 256;        // e_dst = h@W1+b1
    const float* eg = Eg + (size_t)t * 256;
    float sig[8], val[8];
    #pragma unroll
    for (int j = 0; j < 8; j++) {
        int c = lane + 32 * j;
        float g = hs[c] + hd[c] + eg[c];
        float sg = 1.f / (1.f + __expf(-g));
        sig[j] = sg;
        val[j] = sg * Ef[(size_t)t * 256 + c];
    }
    float sm = 0.f;
    #pragma unroll
    for (int j = 0; j < 8; j++) sm += val[j];
    float m = wave_sum32(sm) * (1.f / 256.f);
    float q = 0.f;
    #pragma unroll
    for (int j = 0; j < 8; j++) { float dv = val[j] - m; q += dv * dv; }
    float rstd = rsqrtf(wave_sum32(q) * (1.f / 256.f) + 1e-5f);
    const float* bh = Hout + s * 1024 + 768;        // Bh = h@W4+b4 (dst_update)
    #pragma unroll
    for (int j = 0; j < 8; j++) {
        int c = lane + 32 * j;
        float eo = (val[j] - m) * rstd * lns[c] + lnb[c];
        Ef[(size_t)t * 256 + c]  = eo;
        Eh[(size_t)t * 256 + c]  = f2bf(eo);
        atomicAdd(&ssh[d * 256 + c], bh[c] * sig[j]);
        atomicAdd(&ss[d * 256 + c], sig[j]);
    }
}

// ---------------- EGGC node stage: h_out = LN(Ah + sum_sh/(sum_s+eps)) ----------------
__global__ void __launch_bounds__(256)
k_node(const float* __restrict__ Hout, const float* __restrict__ ssh, const float* __restrict__ ss,
       float* __restrict__ Hf, unsigned short* __restrict__ Hh,
       const float* __restrict__ lns, const float* __restrict__ lnb, int n) {
    int i = blockIdx.x * 8 + (threadIdx.x >> 5);
    if (i >= n) return;
    int lane = threadIdx.x & 31;
    float v[8];
    #pragma unroll
    for (int j = 0; j < 8; j++) {
        int c = lane + 32 * j;
        v[j] = Hout[(long)i * 1024 + 512 + c]       // Ah = h@W3+b3 (src_update)
             + ssh[(long)i * 256 + c] / (ss[(long)i * 256 + c] + 1e-8f);
    }
    float s = 0.f;
    #pragma unroll
    for (int j = 0; j < 8; j++) s += v[j];
    float m = wave_sum32(s) * (1.f / 256.f);
    float q = 0.f;
    #pragma unroll
    for (int j = 0; j < 8; j++) { float d = v[j] - m; q += d * d; }
    float rstd = rsqrtf(wave_sum32(q) * (1.f / 256.f) + 1e-5f);
    #pragma unroll
    for (int j = 0; j < 8; j++) {
        int c = lane + 32 * j;
        float t = (v[j] - m) * rstd * lns[c] + lnb[c];
        Hf[(long)i * 256 + c] = t;
        Hh[(long)i * 256 + c] = f2bf(t);
    }
}

// ---------------- readout ----------------
__global__ void __launch_bounds__(256)
k_pool_acc(const float* __restrict__ X, const int* __restrict__ gid,
           float* __restrict__ pool, float* __restrict__ cnt, int n) {
    int i = blockIdx.x * 8 + (threadIdx.x >> 5);
    if (i >= n) return;
    int lane = threadIdx.x & 31;
    int g = gid[i];
    #pragma unroll
    for (int j = 0; j < 8; j++) {
        int c = lane + 32 * j;
        atomicAdd(&pool[(long)g * 256 + c], X[(long)i * 256 + c]);
    }
    if (lane == 0) atomicAdd(&cnt[g], 1.0f);
}

__global__ void k_pool_fin(const float* __restrict__ pool, const float* __restrict__ cnt,
                           float* __restrict__ out) {
    int i = blockIdx.x * 256 + threadIdx.x;        // 32*256
    int b = i >> 8;
    out[i] = pool[i] / fmaxf(cnt[b], 1.0f);
}

// ---------------- host orchestration ----------------
extern "C" void kernel_launch(void* const* d_in, const int* in_sizes, int n_in,
                              void* d_out, int out_size, void* d_ws, size_t ws_size,
                              hipStream_t stream) {
    (void)in_sizes; (void)n_in; (void)out_size; (void)ws_size;
    const float* atom_features = (const float*)d_in[0];
    const float* bondlength    = (const float*)d_in[1];
    const float* angle_h       = (const float*)d_in[2];
    const int*   src           = (const int*)d_in[3];
    const int*   dst           = (const int*)d_in[4];
    const int*   lg_src        = (const int*)d_in[5];
    const int*   lg_dst        = (const int*)d_in[6];
    const int*   graph_ids     = (const int*)d_in[7];
    const float* atom_W  = (const float*)d_in[8];
    const float* atom_b  = (const float*)d_in[9];
    const float* atom_ln = (const float*)d_in[10];
    const float* edge_W1 = (const float*)d_in[11];
    const float* edge_b1 = (const float*)d_in[12];
    const float* edge_ln1 = (const float*)d_in[13];
    const float* edge_W2 = (const float*)d_in[14];
    const float* edge_b2 = (const float*)d_in[15];
    const float* edge_ln2 = (const float*)d_in[16];
    const float* ang_W1 = (const float*)d_in[17];
    const float* ang_b1 = (const float*)d_in[18];
    const float* ang_ln1 = (const float*)d_in[19];
    const float* ang_W2 = (const float*)d_in[20];
    const float* ang_b2 = (const float*)d_in[21];
    const float* ang_ln2 = (const float*)d_in[22];
    const float* conv_W    = (const float*)d_in[23];
    const float* conv_b    = (const float*)d_in[24];
    const float* conv_ln_s = (const float*)d_in[25];
    const float* conv_ln_b = (const float*)d_in[26];

    // -------- workspace carve (256B aligned) --------
    char* ws = (char*)d_ws;
    size_t off = 0;
    auto take = [&](size_t bytes) -> char* {
        off = (off + 255) & ~(size_t)255;
        char* p = ws + off; off += bytes; return p;
    };
    unsigned short* wcat     = (unsigned short*)take(12ull * 1024 * 256 * 2);
    unsigned short* w2t      = (unsigned short*)take(12ull * 256 * 256 * 2);
    float*          bcat     = (float*)take(12ull * 1024 * 4);
    unsigned short* atom_pad = (unsigned short*)take((size_t)NN * 96 * 2);
    unsigned short* atom_wt  = (unsigned short*)take(256ull * 96 * 2);
    unsigned short* edge_w2t = (unsigned short*)take(256ull * 64 * 2);
    unsigned short* ang_w2t  = (unsigned short*)take(256ull * 64 * 2);
    unsigned short* emb64    = (unsigned short*)take((size_t)TT * 64 * 2);
    float*          x_f = (float*)take((size_t)NN * 256 * 4);
    unsigned short* x_h = (unsigned short*)take((size_t)NN * 256 * 2);
    float*          y_f = (float*)take((size_t)EE * 256 * 4);
    unsigned short* y_h = (unsigned short*)take((size_t)EE * 256 * 2);
    float*          z_f = (float*)take((size_t)TT * 256 * 4);
    unsigned short* z_h = (unsigned short*)take((size_t)TT * 256 * 2);
    float*          hout = (float*)take((size_t)EE * 1024 * 4);
    float*          eg   = (float*)take((size_t)TT * 256 * 4);
    float*          ssh  = (float*)take((size_t)EE * 256 * 4);
    float*          ssum = (float*)take((size_t)EE * 256 * 4);
    float*          pool = (float*)take((8192 + 32) * 4);

    // -------- prep --------
    {
        long total = 12L * 5 * 256 * 256;
        k_prep_convw<<<(unsigned)((total + 255) / 256), 256, 0, stream>>>(conv_W, wcat, w2t);
        k_prep_bcat<<<(12 * 1024 + 255) / 256, 256, 0, stream>>>(conv_b, bcat);
        k_trans_pad<<<(256 * 96 + 255) / 256, 256, 0, stream>>>(atom_W, atom_wt, 92, 96);
        k_trans_pad<<<(256 * 64 + 255) / 256, 256, 0, stream>>>(edge_W2, edge_w2t, 64, 64);
        k_trans_pad<<<(256 * 64 + 255) / 256, 256, 0, stream>>>(ang_W2, ang_w2t, 64, 64);
        k_pad_atoms<<<(unsigned)(((long)NN * 96 + 255) / 256), 256, 0, stream>>>(atom_features, atom_pad, NN);
    }

    // -------- embeddings --------
    // bonds: rbf(0..8,80) -> 64 (LN+SiLU) -> 256 (LN+SiLU) => y
    k_rbf_mlp<<<(EE + 127) / 128, 128, 80 * 64 * 4, stream>>>(
        bondlength, emb64, edge_W1, edge_b1, edge_ln1, EE, 80, 0.0f, 8.0f);
    k_gemm<64><<<dim3((EE + 15) / 16, 1), 256, 0, stream>>>(emb64, edge_w2t, edge_b2, y_f, EE, 256);
    k_ln_act<<<(EE + 7) / 8, 256, 0, stream>>>(y_f, y_h, edge_ln2, edge_ln2 + 256, EE, 1);
    // angles: rbf(-1..1,40) -> 64 -> 256 => z
    k_rbf_mlp<<<(TT + 127) / 128, 128, 40 * 64 * 4, stream>>>(
        angle_h, emb64, ang_W1, ang_b1, ang_ln1, TT, 40, -1.0f, 1.0f);
    k_gemm<64><<<dim3((TT + 15) / 16, 1), 256, 0, stream>>>(emb64, ang_w2t, ang_b2, z_f, TT, 256);
    k_ln_act<<<(TT + 7) / 8, 256, 0, stream>>>(z_f, z_h, ang_ln2, ang_ln2 + 256, TT, 1);
    // atoms: [N,92]->256 => x
    k_gemm<96><<<dim3((NN + 15) / 16, 1), 256, 0, stream>>>(atom_pad, atom_wt, atom_b, x_f, NN, 256);
    k_ln_act<<<(NN + 7) / 8, 256, 0, stream>>>(x_f, x_h, atom_ln, atom_ln + 256, NN, 1);

    // -------- EGGC layers --------
    auto eggc = [&](int layer, const int* es, const int* ed,
                    float* Hf, unsigned short* Hh, int nH,
                    float* Ef, unsigned short* Eh, int nE) {
        const unsigned short* wc = wcat + (size_t)layer * 1024 * 256;
        const unsigned short* w2 = w2t + (size_t)layer * 256 * 256;
        const float* bc  = bcat + (size_t)layer * 1024;
        const float* b2  = conv_b + ((size_t)layer * 5 + 2) * 256;
        const float* lse = conv_ln_s + ((size_t)layer * 2 + 0) * 256;
        const float* lbe = conv_ln_b + ((size_t)layer * 2 + 0) * 256;
        const float* lsn = conv_ln_s + ((size_t)layer * 2 + 1) * 256;
        const float* lbn = conv_ln_b + ((size_t)layer * 2 + 1) * 256;
        k_gemm<256><<<dim3((nH + 15) / 16, 4), 256, 0, stream>>>(Hh, wc, bc, hout, nH, 1024);
        k_gemm<256><<<dim3((nE + 15) / 16, 1), 256, 0, stream>>>(Eh, w2, b2, eg, nE, 256);
        k_zero<<<2048, 256, 0, stream>>>(ssh, (long)nH * 256);
        k_zero<<<2048, 256, 0, stream>>>(ssum, (long)nH * 256);
        k_edge<<<(nE + 7) / 8, 256, 0, stream>>>(es, ed, hout, eg, Ef, Eh, ssh, ssum, lse, lbe, nE);
        k_node<<<(nH + 7) / 8, 256, 0, stream>>>(hout, ssh, ssum, Hf, Hh, lsn, lbn, nH);
    };

    for (int i = 0; i < 4; i++) {
        eggc(2 * i,     lg_src, lg_dst, y_f, y_h, EE, z_f, z_h, TT);  // line-graph update
        eggc(2 * i + 1, src,    dst,    x_f, x_h, NN, y_f, y_h, EE);  // graph update
    }
    for (int i = 0; i < 4; i++)
        eggc(8 + i, src, dst, x_f, x_h, NN, y_f, y_h, EE);

    // -------- readout: per-graph mean --------
    k_zero<<<64, 256, 0, stream>>>(pool, 8192 + 32);
    k_pool_acc<<<(NN + 7) / 8, 256, 0, stream>>>(x_f, graph_ids, pool, pool + 8192, NN);
    k_pool_fin<<<32, 256, 0, stream>>>(pool, pool + 8192, (float*)d_out);
}
